// MolecularGNN_17669495455818
// MI455X (gfx1250) — compile-verified, matrix operand order checked by reading, not run on hardware
//
#include <hip/hip_runtime.h>
#include <hip/hip_bf16.h>

typedef __attribute__((ext_vector_type(16))) __bf16 v16bf;
typedef __attribute__((ext_vector_type(8)))  float  v8f;

#define BN_EPS 1e-5f

// ---------------------------------------------------------------- utilities
__global__ void fill_f32(float* __restrict__ p, int n, float v) {
  int i = blockIdx.x * blockDim.x + threadIdx.x;
  if (i < n) p[i] = v;
}

// ------------------------------------------------------------ degree / dinv
__global__ void edge_deg(const int* __restrict__ dst, float* __restrict__ deg, int E) {
  int i = blockIdx.x * blockDim.x + threadIdx.x;
  if (i < E) atomicAdd(&deg[dst[i]], 1.0f);
}

__global__ void deg_to_dinv(float* __restrict__ d, int n) {
  int i = blockIdx.x * blockDim.x + threadIdx.x;
  if (i < n) d[i] = rsqrtf(d[i]);
}

// -------------------------------------------------- bf16 input prep (layer1)
__global__ void pad_x_bf16(const float* __restrict__ x, __bf16* __restrict__ xp,
                           int N, int Fin, int Kpad) {
  int i = blockIdx.x * blockDim.x + threadIdx.x;
  if (i >= N * Kpad) return;
  int n = i / Kpad, j = i - n * Kpad;
  xp[i] = (j < Fin) ? (__bf16)x[(size_t)n * Fin + j] : (__bf16)0.0f;
}

// ------------------------------------------- pre-swizzle W[KxH] into B-frags
// frag index = ((c*tiles + t)*32 + lane)*16 + e
// lane: n = t*16 + (lane&15), half = lane>>4
// element e -> k = c*32 + ((e<8)? e : e+8) + half*8   (ISA 16-bit B layout)
__global__ void pack_wfrag(const float* __restrict__ W, __bf16* __restrict__ frag,
                           int K, int H, int chunks) {
  int i = blockIdx.x * blockDim.x + threadIdx.x;
  int tiles = H >> 4;
  int total = chunks * tiles * 32 * 16;
  if (i >= total) return;
  int e    = i & 15;
  int lane = (i >> 4) & 31;
  int t    = (i >> 9) % tiles;
  int c    = (i >> 9) / tiles;
  int k = c * 32 + ((e < 8) ? e : e + 8) + (lane >> 4) * 8;
  int n = t * 16 + (lane & 15);
  frag[i] = (k < K) ? (__bf16)W[(size_t)k * H + n] : (__bf16)0.0f;
}

// ----------------------------------------------------------- WMMA bf16 GEMM
// C[N x H] = A[N x Kpad] * W, W already swizzled into B-fragments.
// One block = 16 rows; wave w handles columns [16w, 16w+16).
__global__ __launch_bounds__(256) void gemm_wmma_bf16(
    const __bf16* __restrict__ A, const __bf16* __restrict__ Bfrag,
    float* __restrict__ C, int N, int Kpad, int H, int chunks) {
  const int lane  = threadIdx.x & 31;
  const int wave  = threadIdx.x >> 5;       // column tile
  const int half  = lane >> 4;
  const int l15   = lane & 15;
  const int tiles = H >> 4;
  const int rowBase = blockIdx.x << 4;

  int m = rowBase + l15;
  if (m >= N) m = N - 1;                    // clamp: keep EXEC all-ones
  const __bf16* Arow = A + (size_t)m * Kpad;

  v8f acc = {0.f, 0.f, 0.f, 0.f, 0.f, 0.f, 0.f, 0.f};

  for (int c = 0; c < chunks; ++c) {
    union Frag { v16bf v; uint4 q[2]; };
    Frag au, bu;
    // A: lane half selects K+8; elements 8..15 live at K+16
    const uint4* ap = reinterpret_cast<const uint4*>(Arow + c * 32 + half * 8);
    au.q[0] = ap[0];
    au.q[1] = ap[2];
    // B: pre-swizzled, 32 contiguous bytes per lane
    const uint4* bp = reinterpret_cast<const uint4*>(
        Bfrag + (((size_t)c * tiles + wave) * 32 + lane) * 16);
    bu.q[0] = bp[0];
    bu.q[1] = bp[1];
    acc = __builtin_amdgcn_wmma_f32_16x16x32_bf16(
        false, au.v, false, bu.v, (short)0, acc, false, false);
  }

  const int n  = wave * 16 + l15;
  const int mB = rowBase + half * 8;
#pragma unroll
  for (int r = 0; r < 8; ++r) {
    int mm = mB + r;
    if (mm < N) C[(size_t)mm * H + n] = acc[r];
  }
}

// ------------------------------------------------- aggregation (GCN message)
__global__ void init_agg(const float* __restrict__ xw, const float* __restrict__ dinv,
                         const float* __restrict__ bias, float* __restrict__ agg,
                         int N, int H) {
  int i = blockIdx.x * blockDim.x + threadIdx.x;
  if (i >= N * H) return;
  int n = i / H, c = i - n * H;
  float di = dinv[n];
  agg[i] = xw[i] * di * di + bias[c];      // self-loop term + bias
}

__global__ void edge_agg(const int* __restrict__ src, const int* __restrict__ dst,
                         const float* __restrict__ dinv, const float* __restrict__ xw,
                         float* __restrict__ agg, int E, int H) {
  int groups = H >> 2;
  int i = blockIdx.x * blockDim.x + threadIdx.x;
  if (i >= E * groups) return;
  int e = i / groups;
  int h = (i - e * groups) << 2;
  int s = src[e], d = dst[e];
  float coef = dinv[s] * dinv[d];
  const float4 v = *reinterpret_cast<const float4*>(xw + (size_t)s * H + h);
  float* o = agg + (size_t)d * H + h;
  atomicAdd(o + 0, v.x * coef);
  atomicAdd(o + 1, v.y * coef);
  atomicAdd(o + 2, v.z * coef);
  atomicAdd(o + 3, v.w * coef);
}

// ----------------------------------------------------------------- BatchNorm
__global__ void bn_stats(const float* __restrict__ X, float* __restrict__ bsum,
                         float* __restrict__ bsq, int N, int H, int rowsPerBlock) {
  int c  = threadIdx.x;                     // blockDim.x == H
  int r0 = blockIdx.x * rowsPerBlock;
  int r1 = min(r0 + rowsPerBlock, N);
  float s = 0.f, q = 0.f;
  for (int r = r0; r < r1; ++r) {
    float v = X[(size_t)r * H + c];
    s += v;
    q += v * v;
  }
  atomicAdd(&bsum[c], s);
  atomicAdd(&bsq[c], q);
}

__global__ void bn_relu(const float* __restrict__ X, const float* __restrict__ bsum,
                        const float* __restrict__ bsq, const float* __restrict__ g,
                        const float* __restrict__ be, float* __restrict__ outF,
                        __bf16* __restrict__ outB, int N, int H) {
  int i = blockIdx.x * blockDim.x + threadIdx.x;
  if (i >= N * H) return;
  int c = i % H;
  float invN = 1.0f / (float)N;
  float mu  = bsum[c] * invN;
  float var = fmaxf(bsq[c] * invN - mu * mu, 0.0f);
  float v = (X[i] - mu) * rsqrtf(var + BN_EPS) * g[c] + be[c];
  v = fmaxf(v, 0.0f);
  if (outF) outF[i] = v;
  if (outB) outB[i] = (__bf16)v;
}

// ------------------------------------------------------------------- pooling
__global__ void pool_count(const int* __restrict__ batch, float* __restrict__ counts,
                           int N) {
  int i = blockIdx.x * blockDim.x + threadIdx.x;
  if (i < N) atomicAdd(&counts[batch[i]], 1.0f);
}

__global__ void pool_acc(const float* __restrict__ h, const int* __restrict__ batch,
                         float* __restrict__ out, int N, int H) {
  int i = blockIdx.x * blockDim.x + threadIdx.x;
  if (i >= N * H) return;
  int n = i / H, c = i - n * H;
  int b = batch[n];
  float v = h[i];
  float* row = out + (size_t)b * 3 * H;
  atomicAdd(row + 2 * H + c, v);                                   // h_add
  // post-ReLU values are >= 0: float bit pattern order == unsigned order
  atomicMax(reinterpret_cast<unsigned int*>(row + H + c), __float_as_uint(v));
}

__global__ void pool_fin(float* __restrict__ out, const float* __restrict__ counts,
                         int B, int H) {
  int i = blockIdx.x * blockDim.x + threadIdx.x;
  if (i >= B * H) return;
  int b = i / H, c = i - b * H;
  float cnt = fmaxf(counts[b], 1.0f);
  out[(size_t)b * 3 * H + c] = out[(size_t)b * 3 * H + 2 * H + c] / cnt;  // h_mean
}

// ------------------------------------------------------------------- driver
extern "C" void kernel_launch(void* const* d_in, const int* in_sizes, int n_in,
                              void* d_out, int out_size, void* d_ws, size_t ws_size,
                              hipStream_t stream) {
  (void)n_in; (void)ws_size;

  const float* x   = (const float*)d_in[0];
  const int*   ei  = (const int*)d_in[1];
  const int*   bat = (const int*)d_in[2];
  const float* W1  = (const float*)d_in[3];
  const float* b1  = (const float*)d_in[4];
  const float* W2  = (const float*)d_in[5];
  const float* b2  = (const float*)d_in[6];
  const float* g1  = (const float*)d_in[7];
  const float* be1 = (const float*)d_in[8];
  const float* g2  = (const float*)d_in[9];
  const float* be2 = (const float*)d_in[10];

  const int N   = in_sizes[2];
  const int E   = in_sizes[1] / 2;
  const int Fin = in_sizes[0] / N;
  const int H   = in_sizes[4];
  const int B   = out_size / (3 * H);
  const int* src = ei;
  const int* dst = ei + E;

  const int Kpad1   = ((Fin + 31) / 32) * 32;
  const int chunks1 = Kpad1 / 32;
  const int chunks2 = H / 32;
  const int tiles   = H / 16;

  // workspace carve (256-B aligned)
  char* wp = (char*)d_ws;
  auto carve = [&](size_t bytes) -> void* {
    void* r = (void*)wp;
    wp += (bytes + 255) & ~(size_t)255;
    return r;
  };
  float*  dinv   = (float*)carve((size_t)N * 4);
  float*  xw     = (float*)carve((size_t)N * H * 4);
  float*  agg    = (float*)carve((size_t)N * H * 4);
  __bf16* hbf    = (__bf16*)carve((size_t)N * H * 2);  // also holds padded x
  __bf16* w1frag = (__bf16*)carve((size_t)chunks1 * tiles * 512 * 2);
  __bf16* w2frag = (__bf16*)carve((size_t)chunks2 * tiles * 512 * 2);
  float*  bsum   = (float*)carve((size_t)H * 4);
  float*  bsq    = (float*)carve((size_t)H * 4);
  float*  counts = (float*)carve((size_t)B * 4);

  const int T = 256;
  auto gsz = [&](long long n) { return (unsigned)((n + T - 1) / T); };
  const int RPB = 512;  // rows per bn_stats block

  // degree -> dinv (self-loop included: init 1.0)
  fill_f32<<<gsz(N), T, 0, stream>>>(dinv, N, 1.0f);
  edge_deg<<<gsz(E), T, 0, stream>>>(dst, dinv, E);
  deg_to_dinv<<<gsz(N), T, 0, stream>>>(dinv, N);

  // pre-swizzle weights into WMMA B-fragment layout (bf16)
  pack_wfrag<<<gsz((long long)chunks1 * tiles * 512), T, 0, stream>>>(W1, w1frag, Fin, H, chunks1);
  pack_wfrag<<<gsz((long long)chunks2 * tiles * 512), T, 0, stream>>>(W2, w2frag, H, H, chunks2);

  // ---------------- layer 1 ----------------
  __bf16* xpad = hbf;  // N x Kpad1 fits inside N x H bf16 buffer
  pad_x_bf16<<<gsz((long long)N * Kpad1), T, 0, stream>>>(x, xpad, N, Fin, Kpad1);
  gemm_wmma_bf16<<<(N + 15) / 16, tiles * 32, 0, stream>>>(xpad, w1frag, xw, N, Kpad1, H, chunks1);
  init_agg<<<gsz((long long)N * H), T, 0, stream>>>(xw, dinv, b1, agg, N, H);
  edge_agg<<<gsz((long long)E * (H / 4)), T, 0, stream>>>(src, dst, dinv, xw, agg, E, H);
  fill_f32<<<gsz(H), T, 0, stream>>>(bsum, H, 0.0f);
  fill_f32<<<gsz(H), T, 0, stream>>>(bsq, H, 0.0f);
  bn_stats<<<(N + RPB - 1) / RPB, H, 0, stream>>>(agg, bsum, bsq, N, H, RPB);
  bn_relu<<<gsz((long long)N * H), T, 0, stream>>>(agg, bsum, bsq, g1, be1, nullptr, hbf, N, H);

  // ---------------- layer 2 ----------------
  gemm_wmma_bf16<<<(N + 15) / 16, tiles * 32, 0, stream>>>(hbf, w2frag, xw, N, H, H, chunks2);
  init_agg<<<gsz((long long)N * H), T, 0, stream>>>(xw, dinv, b2, agg, N, H);
  edge_agg<<<gsz((long long)E * (H / 4)), T, 0, stream>>>(src, dst, dinv, xw, agg, E, H);
  fill_f32<<<gsz(H), T, 0, stream>>>(bsum, H, 0.0f);
  fill_f32<<<gsz(H), T, 0, stream>>>(bsq, H, 0.0f);
  bn_stats<<<(N + RPB - 1) / RPB, H, 0, stream>>>(agg, bsum, bsq, N, H, RPB);
  // h2 (f32) written into xw buffer (no longer needed after edge pass)
  bn_relu<<<gsz((long long)N * H), T, 0, stream>>>(agg, bsum, bsq, g2, be2, xw, nullptr, N, H);

  // ---------------- pooling ----------------
  fill_f32<<<gsz((long long)B * 3 * H), T, 0, stream>>>((float*)d_out, B * 3 * H, 0.0f);
  fill_f32<<<gsz(B), T, 0, stream>>>(counts, B, 0.0f);
  pool_count<<<gsz(N), T, 0, stream>>>(bat, counts, N);
  pool_acc<<<gsz((long long)N * H), T, 0, stream>>>(xw, bat, (float*)d_out, N, H);
  pool_fin<<<gsz((long long)B * H), T, 0, stream>>>((float*)d_out, counts, B, H);
}
